// TemporalAttention_65489661330045
// MI455X (gfx1250) — compile-verified, matrix operand order checked by reading, not run on hardware
//
#include <hip/hip_runtime.h>
#include <hip/hip_bf16.h>

// Problem constants (from reference): B=4, C=512, T=24, H=W=32, NH=8, GROUPS=32
#define CDIM 512
#define TDIM 24
#define TPAD 32
#define NHEADS 8
#define HDIM 64      // CDIM / NHEADS
#define HW 1024      // H*W
#define NPOS 4096    // B*H*W

// LDS layout (byte offsets inside dynamic smem; no static __shared__ in kernel,
// so dynamic smem starts at LDS offset 0 — used as TDM lds_addr base).
#define OFF_X    0          // f32 x slice [C][T]            : 49152 B
#define OFF_B1   49152      // bf16 xn [TPAD][C] / f32 scores: 32768 B
#define OFF_B2   81920      // bf16 q [TPAD][C]  / bf16 probs: 32768 B
#define OFF_B3   114688     // bf16 k [TPAD][C]  / bf16 attnO: 32768 B
#define OFF_B4   147456     // bf16 v [C][TPAD]              : 32768 B
#define OFF_RED  180224     // f32 reduction scratch         : 2304 B
#define OFF_WPAN 182528     // bf16 TDM weight panel         : 98304 B
#define SMEM_TOTAL (OFF_WPAN + 98304)   // 280832 B < 320 KB WGP budget

typedef __attribute__((ext_vector_type(16))) __bf16 bf16x16;
typedef __attribute__((ext_vector_type(8)))  float  f32x8;
typedef __attribute__((ext_vector_type(4)))  unsigned int u32x4;
typedef __attribute__((ext_vector_type(8)))  int    i32x8;
typedef __attribute__((ext_vector_type(4)))  int    i32x4;

// Load one 16x32 (MxK) bf16 WMMA fragment: lane (l&15) selects the row,
// (l>>4) selects which 16-element half of K the lane carries. Contiguous
// 32-byte vector load per lane (ds_load_b128 pairs).
__device__ __forceinline__ bf16x16 frag_ld(const __bf16* p0, int rowBase, int ld, int colBase) {
    const int lane = threadIdx.x & 31;
    const __bf16* p = p0 + (size_t)(rowBase + (lane & 15)) * ld + colBase + ((lane >> 4) << 4);
    return *(const bf16x16*)p;
}

// Issue a 2D TENSOR_LOAD_TO_LDS: tile_d1 rows of tile_d0 elements (row stride
// stride_d0 elements of size 2^ds_log bytes) from gaddr into LDS at lds_off.
// Descriptor bitfields per CDNA5 ISA.
__device__ __forceinline__ void tdm_load_2d(const void* gaddr, unsigned lds_off,
                                            unsigned tile_d0, unsigned tile_d1,
                                            unsigned tensor_d0, unsigned tensor_d1,
                                            unsigned stride_d0, unsigned ds_log) {
    unsigned long long ga = (unsigned long long)gaddr;
    u32x4 g0;
    g0[0] = 1u;                                            // count=1, user mode
    g0[1] = lds_off;                                       // lds_addr (bytes)
    g0[2] = (unsigned)(ga & 0xffffffffu);                  // global_addr[31:0]
    g0[3] = (unsigned)((ga >> 32) & 0x1ffffffu) | (2u << 30); // ga[56:32], type=2
    i32x8 g1;
    g1[0] = (int)(ds_log << 16);                           // wg_mask=0, data_size
    g1[1] = (int)((tensor_d0 & 0xffffu) << 16);            // abar=0 | tdim0 lo16
    g1[2] = (int)((tensor_d0 >> 16) | ((tensor_d1 & 0xffffu) << 16));
    g1[3] = (int)((tensor_d1 >> 16) | (tile_d0 << 16));    // tdim1 hi | tile_dim0
    g1[4] = (int)(tile_d1 & 0xffffu);                      // tile_dim1, tile_dim2=0
    g1[5] = (int)stride_d0;                                // dim0 stride lo32
    g1[6] = 0;                                             // stride hi | dim1 stride
    g1[7] = 0;
    i32x4 gz = {0, 0, 0, 0};
#if defined(__clang_major__) && (__clang_major__ >= 23)
    i32x8 gz8 = {0, 0, 0, 0, 0, 0, 0, 0};
    __builtin_amdgcn_tensor_load_to_lds(g0, g1, gz, gz, gz8, 0);
#else
    __builtin_amdgcn_tensor_load_to_lds(g0, g1, gz, gz, 0);
#endif
}

// ---------------------------------------------------------------------------
// Kernel 0: stage weights to bf16 in workspace (read once, live in L2 after)
// ---------------------------------------------------------------------------
__global__ void prep_weights(const float* __restrict__ qkv_w,
                             const float* __restrict__ proj_w,
                             __bf16* __restrict__ wq,
                             __bf16* __restrict__ wp) {
    int i = blockIdx.x * blockDim.x + threadIdx.x;
    if (i < 3 * CDIM * CDIM) {
        wq[i] = (__bf16)qkv_w[i];
    } else {
        int j = i - 3 * CDIM * CDIM;
        if (j < CDIM * CDIM) wp[j] = (__bf16)proj_w[j];
    }
}

// ---------------------------------------------------------------------------
// Kernel 1: fused GroupNorm + QKV + attention + proj + residual, one WG per n
// ---------------------------------------------------------------------------
__global__ __launch_bounds__(256)
void ta_fused(const float* __restrict__ x,
              const float* __restrict__ gn_w, const float* __restrict__ gn_b,
              const __bf16* __restrict__ wq,  const float* __restrict__ qkv_b,
              const __bf16* __restrict__ wp,  const float* __restrict__ proj_b,
              float* __restrict__ out) {
    extern __shared__ __align__(32) char smem[];
    float*  sX    = (float*)(smem + OFF_X);    // x slice, live all kernel (residual)
    __bf16* sXnT  = (__bf16*)(smem + OFF_B1);  // normalized x, [TPAD][C]
    float*  sS    = (float*)(smem + OFF_B1);   // reused: f32 scores [NH][TPAD][TPAD]
    __bf16* sQt   = (__bf16*)(smem + OFF_B2);  // q, [TPAD][C]
    __bf16* sP    = (__bf16*)(smem + OFF_B2);  // reused: probs [NH][TPAD][TPAD]
    __bf16* sKt   = (__bf16*)(smem + OFF_B3);  // k, [TPAD][C]
    __bf16* sO    = (__bf16*)(smem + OFF_B3);  // reused: attn out [TPAD][C]
    __bf16* sV    = (__bf16*)(smem + OFF_B4);  // v, [C][TPAD]
    float*  sRed  = (float*)(smem + OFF_RED);  // 512 f32 partials
    float*  sStat = sRed + 512;                // 64 f32 (mean | rstd)
    __bf16* sWpan = (__bf16*)(smem + OFF_WPAN);// TDM-staged weight panel

    const int tid  = threadIdx.x;
    const int lane = tid & 31;
    const int wave = tid >> 5;
    const int n    = blockIdx.x;
    const int bb   = n >> 10;      // batch
    const int hw   = n & 1023;     // h*W + w

    // ---- Phase A: gather x slice (C x T) into LDS ----
    // 12288 independent 4-byte loads with 4KB stride: parallel per-lane VMEM
    // gather beats a serial TDM walk here.
    for (int i = tid; i < CDIM * TDIM; i += 256) {
        int c = i / TDIM, t = i - c * TDIM;
        sX[i] = x[((size_t)(bb * CDIM + c) * TDIM + t) * HW + hw];
    }
    __syncthreads();

    // ---- Phase B: GroupNorm (32 groups of 16 channels x 24 t) ----
    {
        int g = tid >> 3, sub = tid & 7;   // 8 threads per group
        float s0 = 0.f, s1 = 0.f;
        for (int cc = 0; cc < 2; ++cc) {
            const float* row = sX + (g * 16 + sub * 2 + cc) * TDIM;
            for (int t = 0; t < TDIM; ++t) { float v = row[t]; s0 += v; s1 += v * v; }
        }
        sRed[tid] = s0; sRed[256 + tid] = s1;
    }
    __syncthreads();
    if ((tid & 7) == 0) {
        int g = tid >> 3;
        float a = 0.f, q2 = 0.f;
        for (int k = 0; k < 8; ++k) { a += sRed[g * 8 + k]; q2 += sRed[256 + g * 8 + k]; }
        float mean = a * (1.f / 384.f);
        float var  = q2 * (1.f / 384.f) - mean * mean;
        sStat[g] = mean;
        sStat[32 + g] = rsqrtf(var + 1e-5f);
    }
    __syncthreads();
    // normalize -> bf16, t-major [TPAD][C], zero-pad t = 24..31
    for (int i = tid; i < CDIM * TDIM; i += 256) {
        int c = i / TDIM, t = i - c * TDIM;
        int g = c >> 4;
        float v = (sX[i] - sStat[g]) * sStat[32 + g] * gn_w[c] + gn_b[c];
        sXnT[t * CDIM + c] = (__bf16)v;
    }
    for (int i = tid; i < CDIM * (TPAD - TDIM); i += 256) {
        int tp = TDIM + (i >> 9);
        int c  = i & 511;
        sXnT[tp * CDIM + c] = (__bf16)0.f;
    }
    __syncthreads();

    // ---- Phase C: QKV GEMM (1536 x 512) x (512 x TPAD) ----
    // 16 M-blocks of 96 rows; TDM stages each 96x512 bf16 A-panel into LDS,
    // all 8 waves consume it. 12 tiles (6 M-tiles x 2 N-tiles) per block.
    const float qscale = 0.125f;   // 1/sqrt(64), folded into q
    for (int mb = 0; mb < 16; ++mb) {
        if (wave == 0) {
            tdm_load_2d((const char*)wq + (size_t)mb * 96 * CDIM * 2, OFF_WPAN,
                        /*tile*/ CDIM, 96, /*tensor*/ CDIM, 3 * CDIM,
                        /*stride*/ CDIM, /*2B*/ 1);
            __builtin_amdgcn_s_wait_tensorcnt((short)0);
        }
        __syncthreads();
        for (int tile = wave; tile < 12; tile += 8) {
            int ml = tile >> 1, nn = tile & 1;
            f32x8 acc = {};
            #pragma unroll 4
            for (int k0 = 0; k0 < CDIM; k0 += 32) {
                bf16x16 av = frag_ld(sWpan, ml * 16, CDIM, k0);
                bf16x16 bv = frag_ld(sXnT, nn * 16, CDIM, k0);
                acc = __builtin_amdgcn_wmma_f32_16x16x32_bf16(false, av, false, bv,
                                                              (short)0, acc, false, false);
            }
            // dBase is tile-uniform; q/k/v boundaries are 16-aligned, so the
            // whole tile goes to exactly one destination (no divergent stores).
            int dBase  = mb * 96 + ml * 16;
            int t      = nn * 16 + (lane & 15);
            int rowOff = (lane >> 4) << 3;
            if (dBase < 512) {
                #pragma unroll
                for (int r = 0; r < 8; ++r) {
                    int d = dBase + r + rowOff;
                    sQt[t * CDIM + d] = (__bf16)((acc[r] + qkv_b[d]) * qscale);
                }
            } else if (dBase < 1024) {
                #pragma unroll
                for (int r = 0; r < 8; ++r) {
                    int d = dBase + r + rowOff;
                    sKt[t * CDIM + (d - 512)] = (__bf16)(acc[r] + qkv_b[d]);
                }
            } else {
                #pragma unroll
                for (int r = 0; r < 8; ++r) {
                    int d = dBase + r + rowOff;
                    sV[(d - 1024) * TPAD + t] = (__bf16)(acc[r] + qkv_b[d]);
                }
            }
        }
        __syncthreads();
    }

    // ---- Phase D: scores S = q^T k per head (K = 64 -> 2 WMMA k-steps) ----
    // 8 heads x 2 mt x 2 nt = 32 tiles -> 4/wave
    for (int j = 0; j < 4; ++j) {
        int tile = wave * 4 + j;
        int h = tile >> 2, rem = tile & 3;
        int mt = rem >> 1, nt = rem & 1;
        f32x8 acc = {};
        #pragma unroll
        for (int k0 = 0; k0 < HDIM; k0 += 32) {
            bf16x16 av = frag_ld(sQt, mt * 16, CDIM, h * HDIM + k0);
            bf16x16 bv = frag_ld(sKt, nt * 16, CDIM, h * HDIM + k0);
            acc = __builtin_amdgcn_wmma_f32_16x16x32_bf16(false, av, false, bv,
                                                          (short)0, acc, false, false);
        }
        int s = nt * 16 + (lane & 15);
        int rowOff = (lane >> 4) << 3;
        #pragma unroll
        for (int r = 0; r < 8; ++r) {
            int t = mt * 16 + r + rowOff;
            sS[(h * TPAD + t) * TPAD + s] = acc[r];
        }
    }
    __syncthreads();

    // ---- softmax over s (f32), re-quantize probs to bf16 with zero padding ----
    if (tid < NHEADS * TDIM) {
        int h = tid / TDIM, t = tid - h * TDIM;
        const float* row = sS + (h * TPAD + t) * TPAD;
        float mx = -1e30f;
        #pragma unroll
        for (int s = 0; s < TDIM; ++s) mx = fmaxf(mx, row[s]);
        float e[TDIM];
        float sum = 0.f;
        #pragma unroll
        for (int s = 0; s < TDIM; ++s) { e[s] = __expf(row[s] - mx); sum += e[s]; }
        float inv = 1.f / sum;
        __bf16* prow = sP + (h * TPAD + t) * TPAD;
        #pragma unroll
        for (int s = 0; s < TDIM; ++s) prow[s] = (__bf16)(e[s] * inv);
        #pragma unroll
        for (int s = TDIM; s < TPAD; ++s) prow[s] = (__bf16)0.f;
    } else {
        int idx = tid - NHEADS * TDIM;           // 0..63: zero pad rows t=24..31
        int h = idx >> 3, t = TDIM + (idx & 7);
        __bf16* prow = sP + (h * TPAD + t) * TPAD;
        #pragma unroll
        for (int s = 0; s < TPAD; ++s) prow[s] = (__bf16)0.f;
    }
    __syncthreads();

    // ---- Phase E: out = P x V per head (K = TPAD = 32, single WMMA step) ----
    // 8 heads x 2 mt x 4 nt = 64 tiles -> 8/wave
    for (int j = 0; j < 8; ++j) {
        int tile = wave * 8 + j;
        int h = tile >> 3, rem = tile & 7;
        int mt = rem >> 2, nt = rem & 3;
        f32x8 acc = {};
        bf16x16 av = frag_ld(sP, h * TPAD + mt * 16, TPAD, 0);
        bf16x16 bv = frag_ld(sV, h * HDIM + nt * 16, TPAD, 0);
        acc = __builtin_amdgcn_wmma_f32_16x16x32_bf16(false, av, false, bv,
                                                      (short)0, acc, false, false);
        int d = h * HDIM + nt * 16 + (lane & 15);
        int rowOff = (lane >> 4) << 3;
        #pragma unroll
        for (int r = 0; r < 8; ++r) {
            int t = mt * 16 + r + rowOff;
            sO[t * CDIM + d] = (__bf16)acc[r];
        }
    }
    __syncthreads();

    // ---- Phase F: proj GEMM (512 x 512) x (512 x TPAD) + bias + residual ----
    // K-outer loop: TDM stages each 512x32 bf16 A-panel (32 KB); 8 register-
    // resident accumulator tiles per wave (64 output tiles over 8 waves).
    f32x8 acc[8];
    #pragma unroll
    for (int j = 0; j < 8; ++j) acc[j] = (f32x8){};
    for (int k0 = 0; k0 < CDIM; k0 += 32) {
        if (wave == 0) {
            tdm_load_2d((const char*)wp + (size_t)k0 * 2, OFF_WPAN,
                        /*tile*/ 32, CDIM, /*tensor*/ CDIM, CDIM,
                        /*stride*/ CDIM, /*2B*/ 1);
            __builtin_amdgcn_s_wait_tensorcnt((short)0);
        }
        __syncthreads();
        #pragma unroll
        for (int j = 0; j < 8; ++j) {
            int tile = wave * 8 + j;
            int m = tile >> 1, nn = tile & 1;
            bf16x16 av = frag_ld(sWpan, m * 16, 32, 0);
            bf16x16 bv = frag_ld(sO, nn * 16, CDIM, k0);
            acc[j] = __builtin_amdgcn_wmma_f32_16x16x32_bf16(false, av, false, bv,
                                                             (short)0, acc[j], false, false);
        }
        __syncthreads();
    }
    #pragma unroll
    for (int j = 0; j < 8; ++j) {
        int tile = wave * 8 + j;
        int m = tile >> 1, nn = tile & 1;
        int t = nn * 16 + (lane & 15);
        int rowOff = (lane >> 4) << 3;
        if (t < TDIM) {
            #pragma unroll
            for (int r = 0; r < 8; ++r) {
                int c = m * 16 + r + rowOff;
                float v = acc[j][r] + proj_b[c] + sX[c * TDIM + t];
                out[((size_t)(bb * CDIM + c) * TDIM + t) * HW + hw] = v;
            }
        }
    }
}

extern "C" void kernel_launch(void* const* d_in, const int* in_sizes, int n_in,
                              void* d_out, int out_size, void* d_ws, size_t ws_size,
                              hipStream_t stream) {
    const float* x      = (const float*)d_in[0];
    const float* gn_w   = (const float*)d_in[1];
    const float* gn_b   = (const float*)d_in[2];
    const float* qkv_w  = (const float*)d_in[3];
    const float* qkv_b  = (const float*)d_in[4];
    const float* proj_w = (const float*)d_in[5];
    const float* proj_b = (const float*)d_in[6];
    float* out = (float*)d_out;

    __bf16* wq = (__bf16*)d_ws;
    __bf16* wp = (__bf16*)((char*)d_ws + (size_t)3 * CDIM * CDIM * 2);

    // 3*512*512 + 512*512 = 1,048,576 elements -> 4096 blocks x 256
    prep_weights<<<4096, 256, 0, stream>>>(qkv_w, proj_w, wq, wp);

    const size_t smem = SMEM_TOTAL;
    (void)hipFuncSetAttribute((const void*)ta_fused,
                              hipFuncAttributeMaxDynamicSharedMemorySize, (int)smem);
    ta_fused<<<NPOS, 256, smem, stream>>>(x, gn_w, gn_b, wq, qkv_b, wp, proj_b, out);
}
